// GroupedExperts_25426206392801
// MI455X (gfx1250) — matrix operand order repaired and checked by reading, not compiled, not run on hardware
//
#include <hip/hip_runtime.h>
#include <stdint.h>

// MoE SwiGLU grouped GEMM for gfx1250 (MI455X), bf16 WMMA w/ f32 accumulate.
// Double-buffered LDS, 2-deep software pipeline (1 k-block in LDS, 1 in regs),
// exactly one workgroup barrier per k-step.
//   MODE 0: ws  = bf16(silu(x @ w1[e]^T))          (T x H)
//   MODE 1: ws  = bf16(ws * (x @ w3[e]^T))         (T x H)  -- in-place
//   MODE 2: out = (ws as bf16) @ w2[e]^T  (fp32)   (T x D)

typedef __attribute__((ext_vector_type(16))) __bf16          v16bf;
typedef __attribute__((ext_vector_type(8)))  __bf16          v8bf;
typedef __attribute__((ext_vector_type(8)))  float           v8f;
typedef __attribute__((ext_vector_type(8)))  unsigned short  u16x8;

constexpr int E_ = 8;
constexpr int T_ = 16384;
constexpr int D_ = 2048;
constexpr int H_ = 5632;

constexpr int BM = 128, BN = 128, BK = 32, THREADS = 256;
constexpr int LDSS = 40;  // bf16 elems per LDS row: 32 + 8 pad (80B = 20-bank stride, 16B aligned)

// Hardware bf16 converts: plain casts -> v_cvt_pk_bf16_f32.
__device__ __forceinline__ unsigned short f2bf(float f) {
  union { __bf16 b; unsigned short u; } c;
  c.b = (__bf16)f;
  return c.u;
}
__device__ __forceinline__ float bf2f(unsigned short h) {
  union { unsigned short u; __bf16 b; } c;
  c.u = h;
  return (float)c.b;
}
__device__ __forceinline__ u16x8 pack8(const float4& a, const float4& b) {
  union { v8bf v; u16x8 u; } c;
  c.v[0] = (__bf16)a.x; c.v[1] = (__bf16)a.y; c.v[2] = (__bf16)a.z; c.v[3] = (__bf16)a.w;
  c.v[4] = (__bf16)b.x; c.v[5] = (__bf16)b.y; c.v[6] = (__bf16)b.z; c.v[7] = (__bf16)b.w;
  return c.u;
}

union FragU { u16x8 u[2]; v16bf v; };

// A fragment (16x32 bf16, ISA 7.12.2): lane holds row m = lane%16.
// k pattern: base = 8*(lane/16); elems 0..7 -> k=base+0..7, elems 8..15 -> k=base+16..23.
__device__ __forceinline__ v16bf load_frag_a(const unsigned short* tile, int lane) {
  const int r    = lane & 15;
  const int base = (lane >> 4) << 3;
  const unsigned short* p = tile + r * LDSS + base;
  FragU f;
  f.u[0] = *(const u16x8*)(p);
  f.u[1] = *(const u16x8*)(p + 16);
  return f.v;
}

// B fragment (32x16 bf16): lane holds col n = lane%16; base = 8*(lane/16); elems 0..7 -> k=base+0..7, elems 8..15 -> k=base+16..23.
__device__ __forceinline__ v16bf load_frag_b(const unsigned short* tile, int lane) {
  const int r    = lane & 15;
  const int base = (lane >> 4) << 3;
  const unsigned short* p = tile + r * LDSS + base;
  FragU f;
  f.u[0] = *(const u16x8*)(p);
  f.u[1] = *(const u16x8*)(p + 16);
  return f.v;
}

template <int MODE>
__global__ __launch_bounds__(THREADS)
void moe_wmma_gemm(const float* __restrict__ Afp,          // MODE 0/1: x (T x K) fp32
                   const unsigned short* __restrict__ Abf, // MODE 2:   h (T x K) bf16
                   const float* __restrict__ W,            // weights (E x N x K) fp32, row-major (n,k)
                   const int* __restrict__ ntok,           // (E,)
                   unsigned short* __restrict__ hbuf,      // (T x H) bf16 staging
                   float* __restrict__ out,                // MODE 2: (T x N) fp32
                   int N, int K) {
  __shared__ unsigned short As[2][BM * LDSS];
  __shared__ unsigned short Bs[2][BN * LDSS];

  const int tid  = threadIdx.x;
  const int lane = tid & 31;
  const int wave = tid >> 5;
  const int wm   = wave & 3;   // 32-row band within block tile
  const int wn   = wave >> 2;  // 64-col band within block tile
  const int m0   = blockIdx.y * BM;
  const int n0   = blockIdx.x * BN;

  // Expert owning this token tile (per-expert counts are multiples of BM here,
  // so a tile never straddles an expert boundary).
  int e = 0, start = 0;
  while (e < E_ - 1 && m0 >= start + ntok[e]) { start += ntok[e]; ++e; }
  const float* __restrict__ Wexp = W + (size_t)e * (size_t)N * (size_t)K;

  // Cooperative tile loader mapping: 256 threads, each covers 16 elems of one row.
  const int lrow = tid >> 1;        // 0..127
  const int kh   = (tid & 1) << 4;  // 0 or 16

  const float* pa_f          = Afp + (size_t)(m0 + lrow) * K + kh;  // MODE 0/1
  const unsigned short* pa_h = Abf + (size_t)(m0 + lrow) * K + kh;  // MODE 2
  const float* pb_f          = Wexp + (size_t)(n0 + lrow) * K + kh;

  // ---- pipeline staging registers (raw; converted at LDS-store time) ----
  float4 ra[4], rb[4];
  u16x8  ra16[2];

  auto stage_load = [&](int kb) {
    if (MODE == 2) {
      const u16x8* p = (const u16x8*)(pa_h + kb);
      ra16[0] = p[0];
      ra16[1] = p[1];
      __builtin_prefetch((const char*)(pa_h + kb) + BK * 2, 0, 1);
    } else {
      const float4* p = (const float4*)(pa_f + kb);
      ra[0] = p[0]; ra[1] = p[1]; ra[2] = p[2]; ra[3] = p[3];
      __builtin_prefetch((const char*)(pa_f + kb) + BK * 4, 0, 1);
    }
    const float4* q = (const float4*)(pb_f + kb);
    rb[0] = q[0]; rb[1] = q[1]; rb[2] = q[2]; rb[3] = q[3];
    __builtin_prefetch((const char*)(pb_f + kb) + BK * 4, 0, 1);
  };

  auto stage_store = [&](int buf) {
    u16x8 aLo, aHi;
    if (MODE == 2) {
      aLo = ra16[0]; aHi = ra16[1];
    } else {
      aLo = pack8(ra[0], ra[1]); aHi = pack8(ra[2], ra[3]);
    }
    const u16x8 bLo = pack8(rb[0], rb[1]);
    const u16x8 bHi = pack8(rb[2], rb[3]);
    *(u16x8*)&As[buf][lrow * LDSS + kh]     = aLo;
    *(u16x8*)&As[buf][lrow * LDSS + kh + 8] = aHi;
    *(u16x8*)&Bs[buf][lrow * LDSS + kh]     = bLo;
    *(u16x8*)&Bs[buf][lrow * LDSS + kh + 8] = bHi;
  };

  v8f acc[2][4];
#pragma unroll
  for (int i = 0; i < 2; ++i)
#pragma unroll
    for (int j = 0; j < 4; ++j)
#pragma unroll
      for (int r = 0; r < 8; ++r) acc[i][j][r] = 0.0f;

  const int nsteps = K / BK;

  // ---- prologue: block 0 -> LDS buf 0; block 1 -> regs ----
  stage_load(0);
  stage_store(0);
  if (nsteps > 1) stage_load(BK);

  for (int s = 0; s < nsteps; ++s) {
    __syncthreads();  // buf[cur] stores visible; prev reads of buf[cur^1] done
    const int cur = s & 1;

    if (s + 1 < nsteps) {
      stage_store(cur ^ 1);                       // block s+1 regs -> other buffer
      if (s + 2 < nsteps) stage_load((s + 2) * BK);  // issue block s+2 global loads
    }

    v16bf afr[2], bfr[4];
    afr[0] = load_frag_a(&As[cur][(wm * 32 + 0)  * LDSS], lane);
    afr[1] = load_frag_a(&As[cur][(wm * 32 + 16) * LDSS], lane);
#pragma unroll
    for (int j = 0; j < 4; ++j)
      bfr[j] = load_frag_b(&Bs[cur][(wn * 64 + j * 16) * LDSS], lane);

#pragma unroll
    for (int i = 0; i < 2; ++i)
#pragma unroll
      for (int j = 0; j < 4; ++j)
        acc[i][j] = __builtin_amdgcn_wmma_f32_16x16x32_bf16(
            false, afr[i], false, bfr[j], (short)0, acc[i][j], false, false);
  }

  // ---- epilogue: C layout (ISA 7.12.2): elem r -> M = r + 8*(lane/16), N = lane%16
  const int mb = (lane >> 4) << 3;
  const int nn = lane & 15;
#pragma unroll
  for (int i = 0; i < 2; ++i) {
#pragma unroll
    for (int j = 0; j < 4; ++j) {
#pragma unroll
      for (int r = 0; r < 8; ++r) {
        const int row = m0 + wm * 32 + i * 16 + mb + r;
        const int col = n0 + wn * 64 + j * 16 + nn;
        const float c = acc[i][j][r];
        if (MODE == 0) {
          // silu via fast rcp (one v_rcp_f32, not the IEEE div expansion)
          const float s = c * __builtin_amdgcn_rcpf(1.0f + __expf(-c));
          hbuf[(size_t)row * N + col] = f2bf(s);
        } else if (MODE == 1) {
          const size_t idx = (size_t)row * N + col;
          hbuf[idx] = f2bf(bf2f(hbuf[idx]) * c);  // h = silu(gate) * up
        } else {
          out[(size_t)row * N + col] = c;
        }
      }
    }
  }
}

extern "C" void kernel_launch(void* const* d_in, const int* in_sizes, int n_in,
                              void* d_out, int out_size, void* d_ws, size_t ws_size,
                              hipStream_t stream) {
  (void)in_sizes; (void)n_in; (void)out_size; (void)ws_size;
  const float* x  = (const float*)d_in[0];   // (T, D) fp32
  const float* w1 = (const float*)d_in[1];   // (E, H, D) fp32
  const float* w2 = (const float*)d_in[2];   // (E, D, H) fp32
  const float* w3 = (const float*)d_in[3];   // (E, H, D) fp32
  const int*   nt = (const int*)d_in[4];     // (E,)
  float* out = (float*)d_out;                // (T, D) fp32
  unsigned short* hbuf = (unsigned short*)d_ws;  // (T, H) bf16 staging: 184.5 MB

  dim3 blk(THREADS);
  dim3 g1(H_ / BN, T_ / BM);  // (44, 128)
  dim3 g2(D_ / BN, T_ / BM);  // (16, 128)

  // gate: silu(x @ w1^T) -> hbuf
  moe_wmma_gemm<0><<<g1, blk, 0, stream>>>(x, hbuf, w1, nt, hbuf, nullptr, H_, D_);
  // up: hbuf *= (x @ w3^T)  (in-place SwiGLU)
  moe_wmma_gemm<1><<<g1, blk, 0, stream>>>(x, hbuf, w3, nt, hbuf, nullptr, H_, D_);
  // down: out = hbuf @ w2^T
  moe_wmma_gemm<2><<<g2, blk, 0, stream>>>(x, hbuf, w2, nt, hbuf, out, D_, H_);
}
